// AliBiAttention_67826123538974
// MI455X (gfx1250) — matrix operand order repaired and checked by reading, not compiled, not run on hardware
//
#include <hip/hip_runtime.h>

// ---------------------------------------------------------------------------
// ALiBi attention for MI455X (gfx1250, wave32, WMMA 16x16x32 f16 -> f32)
// B=4, S=1024, D=1024, H=16, Hd=64
// ---------------------------------------------------------------------------

typedef _Float16 h16 __attribute__((ext_vector_type(16)));
typedef _Float16 h8  __attribute__((ext_vector_type(8)));
typedef float    f8  __attribute__((ext_vector_type(8)));

#define B_  4
#define S_  1024
#define D_  1024
#define H_  16
#define HD_ 64

static __device__ inline h16 cat8(h8 lo, h8 hi) {
  return __builtin_shufflevector(lo, hi, 0,1,2,3,4,5,6,7,8,9,10,11,12,13,14,15);
}

static __device__ inline f8 wmma32(h16 a, h16 b, f8 c) {
  // v_wmma_f32_16x16x32_f16: D = A(16x32) * B(32x16) + C(16x16)
  return __builtin_amdgcn_wmma_f32_16x16x32_f16(false, a, false, b, (short)0, c,
                                                false, false);
}

static __device__ inline float wave_max(float v) {
  #pragma unroll
  for (int o = 16; o > 0; o >>= 1) v = fmaxf(v, __shfl_xor(v, o, 32));
  return v;
}
static __device__ inline float wave_sum(float v) {
  #pragma unroll
  for (int o = 16; o > 0; o >>= 1) v += __shfl_xor(v, o, 32);
  return v;
}

// ---------------------------------------------------------------------------
// f32 -> f16 weight conversion
// ---------------------------------------------------------------------------
__global__ void alibi_cvt_f16(const float* __restrict__ x,
                              _Float16* __restrict__ y, int n) {
  int i = blockIdx.x * blockDim.x + threadIdx.x;
  if (i < n) y[i] = (_Float16)x[i];
}

// ---------------------------------------------------------------------------
// Y = X @ W^T  (X: [4096,1024] TIN, W: [1024,1024] f16 row-major)
// MODE 0: Y f16 laid out [B,H,S,Hd]      (Q / K)
// MODE 1: Y f16 laid out [B,H,Hd,S]      (V transposed)
// MODE 2: Y f32 laid out [4096,1024]     (final output)
// Block: 256 threads = 8 waves, each wave computes a 32x32 tile.
// Block tile: 64 rows x 128 cols. Grid: (1024/128, 4096/64).
// ---------------------------------------------------------------------------
template <int MODE, typename TIN>
__global__ __launch_bounds__(256) void alibi_proj_wmma(
    const TIN* __restrict__ X, const _Float16* __restrict__ W,
    void* __restrict__ Yv) {
  const int lane = threadIdx.x & 31;
  const int wave = threadIdx.x >> 5;
  const int g    = lane >> 4;   // half-wave group
  const int ln   = lane & 15;
  const int rowBase = blockIdx.y * 64 + (wave >> 2) * 32;
  const int colBase = blockIdx.x * 128 + (wave & 3) * 32;

  f8 acc[2][2] = {};
  for (int k0 = 0; k0 < D_; k0 += 32) {
    h16 a[2], bfr[2];
    #pragma unroll
    for (int t = 0; t < 2; ++t) {
      // A fragment (ISA 16-bit A layout: halves 0-7 = K g*8.., 8-15 = K 16+g*8..)
      int row = rowBase + t * 16 + ln;
      if constexpr (sizeof(TIN) == 4) {
        const float* p = (const float*)X + row * D_ + k0 + g * 8;
        f8 lo = *(const f8*)p;
        f8 hi = *(const f8*)(p + 16);
        h16 av;
        #pragma unroll
        for (int i = 0; i < 8; ++i) { av[i] = (_Float16)lo[i]; av[8 + i] = (_Float16)hi[i]; }
        a[t] = av;
      } else {
        const _Float16* p = (const _Float16*)X + row * D_ + k0 + g * 8;
        a[t] = cat8(*(const h8*)p, *(const h8*)(p + 16));
      }
      // B fragment: column n = W row 'col', same K layout as A (g*8.., 16+g*8..)
      int col = colBase + t * 16 + ln;
      bfr[t] = cat8(*(const h8*)(W + col * D_ + k0 + g * 8), *(const h8*)(W + col * D_ + k0 + 16 + g * 8));
    }
    #pragma unroll
    for (int ti = 0; ti < 2; ++ti)
      #pragma unroll
      for (int tj = 0; tj < 2; ++tj)
        acc[ti][tj] = wmma32(a[ti], bfr[tj], acc[ti][tj]);
  }

  // D layout: lane holds col n = ln, rows m = g*8 + r
  #pragma unroll
  for (int ti = 0; ti < 2; ++ti) {
    #pragma unroll
    for (int tj = 0; tj < 2; ++tj) {
      #pragma unroll
      for (int r = 0; r < 8; ++r) {
        int row = rowBase + ti * 16 + g * 8 + r;
        int col = colBase + tj * 16 + ln;
        float v = acc[ti][tj][r];
        if constexpr (MODE == 2) {
          ((float*)Yv)[row * D_ + col] = v;
        } else {
          int bb = row >> 10, s = row & (S_ - 1);
          int hh = col >> 6, hd = col & (HD_ - 1);
          _Float16* Y = (_Float16*)Yv;
          if constexpr (MODE == 0)
            Y[((bb * H_ + hh) * S_ + s) * HD_ + hd] = (_Float16)v;
          else
            Y[((bb * H_ + hh) * HD_ + hd) * S_ + s] = (_Float16)v;
        }
      }
    }
  }
}

// ---------------------------------------------------------------------------
// Attention: one workgroup per (b, h, 16-query block).
//   stage1: scores[16][1024] = Q Kt + slope*min(j-i,0) + mask   (WMMA, ->LDS)
//   stage2: row softmax (wave32 shuffle reductions), P f16 -> LDS
//   stage3: ctx = P @ V via WMMA split-K, deterministic combine, f16 out
// LDS: 64KB scores + 32KB P + 8KB partials + 64B invSum
// ---------------------------------------------------------------------------
__global__ __launch_bounds__(256) void alibi_attn_wmma(
    const _Float16* __restrict__ Q, const _Float16* __restrict__ K,
    const _Float16* __restrict__ Vt, const int* __restrict__ mask,
    _Float16* __restrict__ Ctx) {
  extern __shared__ char smem[];
  float*    scores = (float*)smem;                                  // [16][1024]
  _Float16* P      = (_Float16*)(smem + 16 * S_ * 4);               // [16][1024]
  float*    part   = (float*)(smem + 16 * S_ * 4 + 16 * S_ * 2);    // [2][16][64]
  float*    invSum = (float*)(smem + 16 * S_ * 4 + 16 * S_ * 2 + 2 * 16 * HD_ * 4);

  const int lane = threadIdx.x & 31;
  const int wave = threadIdx.x >> 5;
  const int g    = lane >> 4;
  const int ln   = lane & 15;
  const int q0   = blockIdx.x * 16;
  const int h    = blockIdx.y;
  const int b    = blockIdx.z;
  const float slope = exp2f(-0.5f * (float)(h + 1));  // 2^(-8/16*(h+1)), H=16 pow2

  // ---- stage 1: scores = Q K^T + alibi + mask --------------------------------
  const _Float16* Qb = Q + ((b * H_ + h) * S_ + q0) * HD_;
  const _Float16* Kb = K + (b * H_ + h) * S_ * HD_;
  // A fragments for this 16-row Q tile (Hd=64 -> two K=32 steps)
  h16 a0 = cat8(*(const h8*)(Qb + ln * HD_ +      g * 8),
                *(const h8*)(Qb + ln * HD_ + 16 + g * 8));
  h16 a1 = cat8(*(const h8*)(Qb + ln * HD_ + 32 + g * 8),
                *(const h8*)(Qb + ln * HD_ + 48 + g * 8));
  for (int t = 0; t < 8; ++t) {
    int kt = wave + t * 8;          // key tile (64 tiles over 8 waves)
    int j  = kt * 16 + ln;          // absolute key index for this lane's column
    h16 b0 = cat8(*(const h8*)(Kb + j * HD_ +      g * 8), *(const h8*)(Kb + j * HD_ + 16 + g * 8));
    h16 b1 = cat8(*(const h8*)(Kb + j * HD_ + 32 + g * 8), *(const h8*)(Kb + j * HD_ + 48 + g * 8));
    f8 c = {};
    c = wmma32(a0, b0, c);
    c = wmma32(a1, b1, c);
    float maskAdd = (mask[b * S_ + j] == 0) ? -1e9f : 0.f;
    #pragma unroll
    for (int r = 0; r < 8; ++r) {
      int m = g * 8 + r;            // query row within tile
      int i = q0 + m;               // absolute query index
      float bias = slope * fminf((float)(j - i), 0.f);  // tril(j-i), 0 above diag
      scores[m * S_ + j] = c[r] + bias + maskAdd;
    }
  }
  __syncthreads();

  // ---- stage 2: softmax per row, store unnormalized exp as f16 ---------------
  #pragma unroll
  for (int rr = 0; rr < 2; ++rr) {
    int m = wave * 2 + rr;
    const float* row = scores + m * S_;
    float mx = -3.4e38f;
    for (int k = lane; k < S_; k += 32) mx = fmaxf(mx, row[k]);
    mx = wave_max(mx);
    float sum = 0.f;
    _Float16* prow = P + m * S_;
    for (int k = lane; k < S_; k += 32) {
      float e = __expf(row[k] - mx);
      prow[k] = (_Float16)e;
      sum += e;
    }
    sum = wave_sum(sum);
    if (lane == 0) invSum[m] = 1.f / sum;
  }
  __syncthreads();

  // ---- stage 3: ctx = P @ V  (split-K: 4 hd-tiles x 2 K-halves over 8 waves) -
  const int hdt   = wave & 3;
  const int khalf = wave >> 2;
  const _Float16* Vb = Vt + ((b * H_ + h) * HD_ + hdt * 16 + ln) * S_;
  f8 c3 = {};
  for (int st = 0; st < 16; ++st) {
    int kbase = khalf * 512 + st * 32;
    const _Float16* pp = P + ln * S_ + kbase + g * 8;
    h16 a  = cat8(*(const h8*)pp, *(const h8*)(pp + 16));
    h16 bb = cat8(*(const h8*)(Vb + kbase + g * 8), *(const h8*)(Vb + kbase + 16 + g * 8));
    c3 = wmma32(a, bb, c3);
  }
  // deterministic combine: each LDS slot written by exactly one lane
  #pragma unroll
  for (int r = 0; r < 8; ++r) {
    int m = g * 8 + r;
    part[(khalf * 16 + m) * HD_ + hdt * 16 + ln] = c3[r];
  }
  __syncthreads();

  for (int e = threadIdx.x; e < 16 * HD_; e += 256) {
    int m = e >> 6, hd = e & (HD_ - 1);
    float v = (part[e] + part[16 * HD_ + e]) * invSum[m];
    int s = q0 + m;
    Ctx[(b * S_ + s) * D_ + h * HD_ + hd] = (_Float16)v;
  }
}

// ---------------------------------------------------------------------------
// Launch: cvt weights -> 3 projections -> attention -> output projection
// Workspace layout (bytes):
//   [0,2M)   Wq16   [2M,4M)  Wout16
//   [4M,12M)  Q16 [B,H,S,Hd]     [12M,20M) K16 [B,H,S,Hd]
//   [20M,28M) Vt16 [B,H,Hd,S]    [28M,36M) Ctx16 [B*S, D]
// ---------------------------------------------------------------------------
extern "C" void kernel_launch(void* const* d_in, const int* in_sizes, int n_in,
                              void* d_out, int out_size, void* d_ws, size_t ws_size,
                              hipStream_t stream) {
  const float* q    = (const float*)d_in[0];
  const float* k    = (const float*)d_in[1];
  const float* v    = (const float*)d_in[2];
  const int*   mask = (const int*)d_in[3];
  const float* Wq   = (const float*)d_in[4];
  const float* Wout = (const float*)d_in[5];

  char* ws = (char*)d_ws;
  _Float16* Wq16  = (_Float16*)(ws);
  _Float16* Wo16  = (_Float16*)(ws + (2u << 20));
  _Float16* Q16   = (_Float16*)(ws + (4u << 20));
  _Float16* K16   = (_Float16*)(ws + (12u << 20));
  _Float16* Vt16  = (_Float16*)(ws + (20u << 20));
  _Float16* Ctx16 = (_Float16*)(ws + (28u << 20));

  alibi_cvt_f16<<<4096, 256, 0, stream>>>(Wq,   Wq16, 1 << 20);
  alibi_cvt_f16<<<4096, 256, 0, stream>>>(Wout, Wo16, 1 << 20);

  dim3 pg(D_ / 128, (B_ * S_) / 64);  // (8, 64)
  alibi_proj_wmma<0, float><<<pg, 256, 0, stream>>>(q, Wq16, Q16);
  alibi_proj_wmma<0, float><<<pg, 256, 0, stream>>>(k, Wq16, K16);
  alibi_proj_wmma<1, float><<<pg, 256, 0, stream>>>(v, Wq16, Vt16);

  dim3 ag(S_ / 16, H_, B_);           // (64, 16, 4)
  size_t shmem = 16 * S_ * 4 + 16 * S_ * 2 + 2 * 16 * HD_ * 4 + 16 * 4;
  alibi_attn_wmma<<<ag, 256, shmem, stream>>>(Q16, K16, Vt16, mask, Ctx16);

  alibi_proj_wmma<2, _Float16><<<pg, 256, 0, stream>>>(Ctx16, Wo16, (float*)d_out);
}
